// W4A8QMamba_86672440033341
// MI455X (gfx1250) — compile-verified, matrix operand order checked by reading, not run on hardware
//
#include <hip/hip_runtime.h>
#include <hip/hip_bf16.h>

// ---------------------------------------------------------------------------
// W4A8 quantized Mamba block on gfx1250 (MI455X), wave32 + WMMA f16 +
// async global->LDS staging (double buffered) for the big GEMMs.
// Wave tile 64x32 (8 WMMAs / 6 fragment builds), block tile 128x64, 4 waves.
// ---------------------------------------------------------------------------

typedef __attribute__((ext_vector_type(16))) _Float16 v16h;
typedef __attribute__((ext_vector_type(8)))  float    v8f;
typedef int vi4 __attribute__((vector_size(16)));   // matches builtin prototype

constexpr int BB  = 2;      // batch
constexpr int LL  = 1024;   // sequence
constexpr int DM  = 2048;   // d_model
constexpr int DI  = 4096;   // d_inner
constexpr int E2  = 2 * DI; // 8192 (x and z)
constexpr int DS  = 16;     // d_state
constexpr int DR  = 128;    // dt_rank
constexpr int XD  = DR + 2 * DS; // 160
constexpr int LDSPAD = 36;  // padded row stride (dwords): 16B-aligned, bank-conflict-free

#define AS3 __attribute__((address_space(3)))
#define AS1 __attribute__((address_space(1)))

#if defined(__gfx1250__) && __has_builtin(__builtin_amdgcn_global_load_async_to_lds_b128)
#define HAVE_ASYNC 1
#else
#define HAVE_ASYNC 0
#endif

__device__ __forceinline__ void copy16_g2l(const float* __restrict__ g, float* __restrict__ l) {
#if HAVE_ASYNC
    __builtin_amdgcn_global_load_async_to_lds_b128(
        (AS1 vi4*)const_cast<float*>(g), (AS3 vi4*)l, 0, 0);
#else
    *(float4*)l = *(const float4*)g;       // global_load_b128 + ds_store_b128
#endif
}

__device__ __forceinline__ void wait_async_done() {
#if HAVE_ASYNC
#if __has_builtin(__builtin_amdgcn_s_wait_asynccnt)
    __builtin_amdgcn_s_wait_asynccnt(0);
#else
    asm volatile("s_wait_asynccnt 0" ::: "memory");
#endif
#endif
}

__device__ __forceinline__ float quantf(float x, float inv_s, float s) {
    float r = rintf(x * inv_s);            // round-half-to-even, matches jnp.round
    r = fminf(fmaxf(r, -128.f), 127.f);
    return r * s;
}

__device__ __forceinline__ float siluf(float x) {
    return x * __builtin_amdgcn_rcpf(1.f + __expf(-x));   // v_rcp_f32, no div chain
}

__device__ __forceinline__ float softplusf(float x) {
    return (x > 20.f) ? x : log1pf(__expf(x));
}

// Load a 16x32 (rows x K) f16 fragment from f32 memory (global or LDS).
// ISA layout: lane<16 holds row=lane, K in {0..7} (elems 0..7) and {16..23}
// (elems 8..15); lane>=16 holds row=lane-16, K in {8..15} and {24..31}.
__device__ __forceinline__ v16h load_frag16x32(const float* __restrict__ p0,
                                               int strideK, int strideRow, int lane) {
    int row = lane & 15;
    int kb  = (lane & 16) >> 1; // 0 or 8
    const float* p = p0 + (size_t)row * strideRow;
    v16h f;
#pragma unroll
    for (int j = 0; j < 8; ++j)
        f[j] = (_Float16)p[(size_t)(kb + j) * strideK];
#pragma unroll
    for (int j = 0; j < 8; ++j)
        f[j + 8] = (_Float16)p[(size_t)(16 + kb + j) * strideK];
    return f;
}

// C/D layout: element r of v8f at lane L -> M = r + 8*(L>=16), N = L&15.
__device__ __forceinline__ void store_frag_C(float* __restrict__ p0,
                                             size_t strideM, size_t strideN,
                                             const v8f& c, int lane) {
    int n  = lane & 15;
    int m0 = (lane & 16) >> 1;
#pragma unroll
    for (int r = 0; r < 8; ++r)
        p0[(size_t)(m0 + r) * strideM + (size_t)n * strideN] = c[r];
}

__device__ __forceinline__ v8f wmma_f16(const v16h& a, const v16h& b, const v8f& c) {
    return __builtin_amdgcn_wmma_f32_16x16x32_f16(false, a, false, b,
                                                  (short)0, c, false, false);
}

// ---------------------------------------------------------------------------
// Unified LDS-staged, double-buffered 128x64 block GEMM (4 waves).
// Wave tile 64(M) x 32(N): 8 WMMAs per stage from 4 A-frags + 2 B-frags.
//   C[m,n] = sum_k A[m,k] * B[n,k]     (A: MxK row-major, B rows: strideB)
// MODE 0: in-proj  -> quantize e<DI into out0=xq[b,e,l], else out1=z[b,e,l]
// MODE 1: out-proj -> out0[b,n,m] = c                      (final output)
// MODE 2: dt-proj  -> out0[b,m,n] = softplus(c + bias[m])
// ---------------------------------------------------------------------------
template <int MODE, int KDIM, int STRIDE_B>
__global__ __launch_bounds__(128) void k_gemm_lds(const float* __restrict__ Aw,
                                                  const float* __restrict__ Bact,
                                                  const float* __restrict__ bias,
                                                  float* __restrict__ out0,
                                                  float* __restrict__ out1) {
    __shared__ float sA[2][128][LDSPAD];   // 36 KB
    __shared__ float sB[2][64][LDSPAD];    // 18 KB

    const int b    = blockIdx.z;
    const int tid  = threadIdx.x;
    const int lane = tid & 31;
    const int wave = tid >> 5;              // 0..3
    const int m0   = blockIdx.x * 128;
    const int n0   = blockIdx.y * 64;
    const int wm   = (wave >> 1) * 64;      // wave M offset within block tile
    const int wn   = (wave & 1) * 32;       // wave N offset within block tile
    const float* Bb = Bact + ((size_t)b * LL + n0) * STRIDE_B;

    auto stage = [&](int buf, int k0) {
#pragma unroll
        for (int t = 0; t < 8; ++t) {       // A: 128 rows x 32 f32 = 1024 x 16B
            int c = tid + t * 128;
            int row = c >> 3, kq = c & 7;
            copy16_g2l(Aw + (size_t)(m0 + row) * KDIM + k0 + kq * 4,
                       &sA[buf][row][kq * 4]);
        }
#pragma unroll
        for (int t = 0; t < 4; ++t) {       // B: 64 rows x 32 f32 = 512 x 16B
            int c = tid + t * 128;
            int row = c >> 3, kq = c & 7;
            copy16_g2l(Bb + (size_t)row * STRIDE_B + k0 + kq * 4,
                       &sB[buf][row][kq * 4]);
        }
    };

    v8f acc[4][2];
#pragma unroll
    for (int i = 0; i < 4; ++i)
#pragma unroll
        for (int j = 0; j < 2; ++j) acc[i][j] = {};

    constexpr int NSTAGE = KDIM / 32;
    stage(0, 0);
    for (int s = 0; s < NSTAGE; ++s) {
        wait_async_done();                  // my async writes for buf s&1 landed
        __syncthreads();                    // everyone's landed; prev reads done
        if (s + 1 < NSTAGE) stage((s + 1) & 1, (s + 1) * 32);
        const int buf = s & 1;
        v16h a[4], bf[2];
#pragma unroll
        for (int i = 0; i < 4; ++i)
            a[i]  = load_frag16x32(&sA[buf][wm + 16 * i][0], 1, LDSPAD, lane);
#pragma unroll
        for (int j = 0; j < 2; ++j)
            bf[j] = load_frag16x32(&sB[buf][wn + 16 * j][0], 1, LDSPAD, lane);
#pragma unroll
        for (int i = 0; i < 4; ++i)
#pragma unroll
            for (int j = 0; j < 2; ++j)
                acc[i][j] = wmma_f16(a[i], bf[j], acc[i][j]);
    }

    const int nn = lane & 15;
    const int mo = (lane & 16) >> 1;
#pragma unroll
    for (int i = 0; i < 4; ++i) {
#pragma unroll
        for (int j = 0; j < 2; ++j) {
            const int mB = m0 + wm + 16 * i + mo;
            const int nB = n0 + wn + 16 * j + nn;
            if constexpr (MODE == 0) {
                const bool isX = (m0 < DI);   // 128-row tiles never straddle split
#pragma unroll
                for (int r = 0; r < 8; ++r) {
                    float v = acc[i][j][r];
                    if (isX)
                        out0[((size_t)b * DI + (mB + r)) * LL + nB] = quantf(v, 20.f, 0.05f);
                    else
                        out1[((size_t)b * DI + (mB + r - DI)) * LL + nB] = v;
                }
            } else if constexpr (MODE == 1) {
#pragma unroll
                for (int r = 0; r < 8; ++r)
                    out0[((size_t)b * LL + nB) * DM + (mB + r)] = acc[i][j][r];
            } else {
#pragma unroll
                for (int r = 0; r < 8; ++r) {
                    float v = acc[i][j][r] + bias[mB + r];
                    out0[((size_t)b * DI + (mB + r)) * LL + nB] = softplusf(v);
                }
            }
        }
    }
}

// ---------------------------------------------------------------------------
// Kernel 2: depthwise causal conv(4) + bias + silu + quant  -> u [b,DI,L]
// ---------------------------------------------------------------------------
__global__ __launch_bounds__(256) void k_conv(const float* __restrict__ xq,
                                              const float* __restrict__ cw,
                                              const float* __restrict__ cb,
                                              float* __restrict__ u) {
    size_t idx = (size_t)blockIdx.x * 256 + threadIdx.x;
    if (idx >= (size_t)BB * DI * LL) return;
    int l = (int)(idx % LL);
    int e = (int)((idx / LL) % DI);
    int b = (int)(idx / ((size_t)LL * DI));
    const float* xr = xq + ((size_t)b * DI + e) * LL;
    float s = cb[e];
#pragma unroll
    for (int j = 0; j < 4; ++j) {
        int ls = l - 3 + j;
        if (ls >= 0) s += xr[ls] * cw[e * 4 + j];
    }
    u[idx] = quantf(siluf(s), 20.f, 0.05f);
}

// ---------------------------------------------------------------------------
// Kernel 3: x_dbl[b,l,160] = u^T @ W_x^T   (M=160, N=l, K=DI; B is K-strided)
// one 16x16 output tile per wave, direct global fragments
// ---------------------------------------------------------------------------
__global__ __launch_bounds__(256) void k_gemm_xdbl(const float* __restrict__ u,
                                                   const float* __restrict__ Wx,
                                                   float* __restrict__ xdbl) {
    const int lane = threadIdx.x & 31;
    int wt = blockIdx.x * 8 + (threadIdx.x >> 5);
    const int NT = LL / 16;                 // 64
    int nt = wt % NT;
    int rest = wt / NT;
    int mt = rest % (XD / 16);              // 10
    int b  = rest / (XD / 16);
    const int m0 = mt * 16, n0 = nt * 16;
    const float* ub = u + (size_t)b * DI * LL;

    v8f c = {};
    for (int k = 0; k < DI; k += 32) {
        __builtin_prefetch(Wx + (size_t)m0 * DI + k + 64, 0, 1);
        v16h a  = load_frag16x32(Wx + (size_t)m0 * DI + k, 1, DI, lane);
        // B(k,n) = u[b, k, n0+n]: strideK = LL, strideRow(N) = 1
        v16h bf = load_frag16x32(ub + (size_t)k * LL + n0, LL, 1, lane);
        c = wmma_f16(a, bf, c);
    }
    store_frag_C(xdbl + ((size_t)b * LL + n0) * XD + m0, 1, XD, c, lane);
}

// ---------------------------------------------------------------------------
// Kernel 5: selective scan, fused +D*u and silu(z) gating.
// one thread per (b,d); B_t/C_t staged through LDS in 64-step chunks.
// writes yg[b, l, d] (transposed, ready for FWHT + out-GEMM).
// ---------------------------------------------------------------------------
__global__ __launch_bounds__(256) void k_scan(const float* __restrict__ dt,
                                              const float* __restrict__ u,
                                              const float* __restrict__ xdbl,
                                              const float* __restrict__ zb,
                                              const float* __restrict__ Aarr,
                                              const float* __restrict__ Darr,
                                              float* __restrict__ yg) {
    const int b = blockIdx.y;
    const int d = blockIdx.x * 256 + threadIdx.x;
    __shared__ float sB[64][DS];
    __shared__ float sC[64][DS];

    const float* dtr = dt + ((size_t)b * DI + d) * LL;
    const float* ur  = u  + ((size_t)b * DI + d) * LL;
    const float* zr  = zb + ((size_t)b * DI + d) * LL;
    const float* xb  = xdbl + (size_t)b * LL * XD;

    float Ad[DS];
#pragma unroll
    for (int n = 0; n < DS; ++n) Ad[n] = Aarr[(size_t)d * DS + n];
    const float Dd = Darr[d];

    float h[DS];
#pragma unroll
    for (int n = 0; n < DS; ++n) h[n] = 0.f;

    for (int l0 = 0; l0 < LL; l0 += 64) {
        __syncthreads();
        for (int i = threadIdx.x; i < 64 * 32; i += 256) {
            int ll = i >> 5, c = i & 31;
            float v = xb[(size_t)(l0 + ll) * XD + DR + c];
            if (c < DS) sB[ll][c] = v;
            else        sC[ll][c - DS] = v;
        }
        __syncthreads();
        for (int li = 0; li < 64; ++li) {
            const int l = l0 + li;
            const float dtv = dtr[l];
            const float uv  = ur[l];
            const float du  = dtv * uv;
            float y = 0.f;
#pragma unroll
            for (int n = 0; n < DS; ++n) {
                float dA = __expf(dtv * Ad[n]);
                h[n] = dA * h[n] + du * sB[li][n];
                y += h[n] * sC[li][n];
            }
            y += Dd * uv;
            yg[((size_t)b * LL + l) * DI + d] = y * siluf(zr[l]);
        }
    }
}

// ---------------------------------------------------------------------------
// Kernel 6: in-place normalized FWHT over d_inner (4096) per (b,l) row.
// ---------------------------------------------------------------------------
__global__ __launch_bounds__(256) void k_fwht(float* __restrict__ yg) {
    __shared__ float s[DI];  // 16 KB
    float* p = yg + (size_t)blockIdx.x * DI;
    for (int i = threadIdx.x; i < DI; i += 256) s[i] = p[i];
    __syncthreads();
    for (int h = 1; h < DI; h <<= 1) {
        for (int idx = threadIdx.x; idx < DI / 2; idx += 256) {
            int i = ((idx / h) * (h << 1)) + (idx & (h - 1));
            int j = i + h;
            float a = s[i], t = s[j];
            s[i] = a + t;
            s[j] = a - t;
        }
        __syncthreads();
    }
    const float sc = 0.015625f;  // 4096^-0.5
    for (int i = threadIdx.x; i < DI; i += 256) p[i] = s[i] * sc;
}

// ---------------------------------------------------------------------------
extern "C" void kernel_launch(void* const* d_in, const int* in_sizes, int n_in,
                              void* d_out, int out_size, void* d_ws, size_t ws_size,
                              hipStream_t stream) {
    const float* hs    = (const float*)d_in[0];
    const float* Win   = (const float*)d_in[1];
    const float* convw = (const float*)d_in[2];
    const float* convb = (const float*)d_in[3];
    const float* Wx    = (const float*)d_in[4];
    const float* Wdt   = (const float*)d_in[5];
    const float* dtbia = (const float*)d_in[6];
    const float* Aarr  = (const float*)d_in[7];
    const float* Darr  = (const float*)d_in[8];
    const float* Wout  = (const float*)d_in[9];
    float* out = (float*)d_out;

    float* ws = (float*)d_ws;
    size_t o = 0;
    float* xq   = ws + o; o += (size_t)BB * DI * LL;   // quantized x   [b,e,l]
    float* zb   = ws + o; o += (size_t)BB * DI * LL;   // z             [b,e,l]
    float* u    = ws + o; o += (size_t)BB * DI * LL;   // conv output   [b,e,l]
    float* xdbl = ws + o; o += (size_t)BB * LL * XD;   // dt_in|B|C     [b,l,160]
    float* dt   = ws + o; o += (size_t)BB * DI * LL;   // softplus dt   [b,d,l]
    float* yg   = ws + o; o += (size_t)BB * LL * DI;   // gated y       [b,l,e]
    (void)ws_size; (void)in_sizes; (void)n_in; (void)out_size;

    // 1. input projection (WMMA, async-LDS staged) + quantize/split
    k_gemm_lds<0, DM, DM><<<dim3(E2 / 128, LL / 64, BB), 128, 0, stream>>>(
        Win, hs, nullptr, xq, zb);
    // 2. depthwise conv + silu + quant
    {
        size_t n = (size_t)BB * DI * LL;
        k_conv<<<dim3((unsigned)((n + 255) / 256)), 256, 0, stream>>>(xq, convw, convb, u);
    }
    // 3. x_dbl projection (WMMA)
    k_gemm_xdbl<<<dim3(BB * (XD / 16) * (LL / 16) / 8), 256, 0, stream>>>(u, Wx, xdbl);
    // 4. dt projection + softplus (WMMA, async-LDS staged)
    k_gemm_lds<2, DR, XD><<<dim3(DI / 128, LL / 64, BB), 128, 0, stream>>>(
        Wdt, xdbl, dtbia, dt, nullptr);
    // 5. selective scan + gating
    k_scan<<<dim3(DI / 256, BB), 256, 0, stream>>>(dt, u, xdbl, zb, Aarr, Darr, yg);
    // 6. FWHT over d_inner
    k_fwht<<<dim3(BB * LL), 256, 0, stream>>>(yg);
    // 7. output projection (WMMA, async-LDS staged)
    k_gemm_lds<1, DI, DI><<<dim3(DM / 128, LL / 64, BB), 128, 0, stream>>>(
        Wout, yg, nullptr, out, nullptr);
}
